// ScaleDotProductAttention_70497593197095
// MI455X (gfx1250) — matrix operand order repaired and checked by reading, not compiled, not run on hardware
//
#include <hip/hip_runtime.h>

#define B_ 2
#define H_ 16
#define S_ 2048
#define D_ 64

typedef _Float16 v16h __attribute__((ext_vector_type(16)));
typedef _Float16 v8h  __attribute__((ext_vector_type(8)));
typedef float    v8f  __attribute__((ext_vector_type(8)));

// XOR-lane butterfly within each 16-lane row via DPP16 ROW_XMASK (pure VALU,
// no LDS pipe). Masks 1/2/4/8 never cross the 16-lane row boundary, which is
// exactly the replication domain of our row statistics.
template <int MASK>
__device__ __forceinline__ float dpp_xor(float x) {
    return __int_as_float(__builtin_amdgcn_update_dpp(
        0, __float_as_int(x), 0x160 | MASK, 0xf, 0xf, true));
}
__device__ __forceinline__ float rowmax16(float x) {
    x = fmaxf(x, dpp_xor<1>(x));
    x = fmaxf(x, dpp_xor<2>(x));
    x = fmaxf(x, dpp_xor<4>(x));
    x = fmaxf(x, dpp_xor<8>(x));
    return x;
}
__device__ __forceinline__ float rowsum16(float x) {
    x += dpp_xor<1>(x);
    x += dpp_xor<2>(x);
    x += dpp_xor<4>(x);
    x += dpp_xor<8>(x);
    return x;
}

// Load 8 contiguous f32 and convert into dst[base..base+7] as f16.
__device__ __forceinline__ void load8f_cvt(const float* __restrict__ p, v16h& dst, int base) {
    float4 a = *reinterpret_cast<const float4*>(p);
    float4 b = *reinterpret_cast<const float4*>(p + 4);
    dst[base + 0] = (_Float16)a.x; dst[base + 1] = (_Float16)a.y;
    dst[base + 2] = (_Float16)a.z; dst[base + 3] = (_Float16)a.w;
    dst[base + 4] = (_Float16)b.x; dst[base + 5] = (_Float16)b.y;
    dst[base + 6] = (_Float16)b.z; dst[base + 7] = (_Float16)b.w;
}

// ---------------------------------------------------------------------------
// Pre-pass 1: K (f32) -> Kh (f16, same row-major layout). 8 floats/thread.
// ---------------------------------------------------------------------------
__global__ void k_convert_f16(const float* __restrict__ k, _Float16* __restrict__ kh) {
    const size_t i = ((size_t)blockIdx.x * blockDim.x + threadIdx.x) * 8;
    float4 a = *reinterpret_cast<const float4*>(k + i);
    float4 b = *reinterpret_cast<const float4*>(k + i + 4);
    v8h h;
    h[0] = (_Float16)a.x; h[1] = (_Float16)a.y; h[2] = (_Float16)a.z; h[3] = (_Float16)a.w;
    h[4] = (_Float16)b.x; h[5] = (_Float16)b.y; h[6] = (_Float16)b.z; h[7] = (_Float16)b.w;
    *reinterpret_cast<v8h*>(kh + i) = h;
}

// ---------------------------------------------------------------------------
// Pre-pass 2: V (f32, [bh][s][d]) -> Vt (f16, [bh][d][s]) so the PV B-operand
// (K-dim contiguous per lane) becomes contiguous 32B loads.
// ---------------------------------------------------------------------------
__global__ void v_transpose_f16(const float* __restrict__ v, _Float16* __restrict__ vt) {
    __shared__ float tile[32][33];
    const int bh = blockIdx.z;
    const int s0 = blockIdx.x * 32;
    const int d0 = blockIdx.y * 32;
    const float*    vp  = v  + (size_t)bh * S_ * D_;
    _Float16*       vtp = vt + (size_t)bh * D_ * S_;
    const int tx = threadIdx.x;   // 0..31
    const int ty = threadIdx.y;   // 0..7
#pragma unroll
    for (int j = 0; j < 4; ++j)
        tile[ty + j * 8][tx] = vp[(size_t)(s0 + ty + j * 8) * D_ + d0 + tx];
    __syncthreads();
#pragma unroll
    for (int j = 0; j < 4; ++j)
        vtp[(size_t)(d0 + ty + j * 8) * S_ + s0 + tx] = (_Float16)tile[tx][ty + j * 8];
}

// ---------------------------------------------------------------------------
// One 32-wide KV step of FlashAttention-2 (per 16-row wave strip).
// MASK=true only for the final (diagonal) step.
// ---------------------------------------------------------------------------
template <bool MASK>
__device__ __forceinline__ void fa_step(
    int jc, int qbase, int lane16, int hi,
    const _Float16* __restrict__ kp, const _Float16* __restrict__ vtp,
    const v16h aq[2], v8f o[4], float m[8], float l[8],
    _Float16 (* __restrict__ pb)[40]) {

    const float SC = 0.18033688f;   // (1/sqrt(64)) * log2(e)

    // ---- S = Q K^T : two 16x16 C-tiles over 32 KV columns ----
    v8f s0 = (v8f){}, s1 = (v8f){};
    {
        const _Float16* kr0 = kp + (size_t)(jc + lane16) * D_ + hi * 16;
        const _Float16* kr1 = kr0 + 16 * D_;
        v16h bk;
        bk = *reinterpret_cast<const v16h*>(kr0);
        s0 = __builtin_amdgcn_wmma_f32_16x16x32_f16(false, aq[0], false, bk, (short)0, s0, false, false);
        bk = *reinterpret_cast<const v16h*>(kr0 + 32);
        s0 = __builtin_amdgcn_wmma_f32_16x16x32_f16(false, aq[1], false, bk, (short)0, s0, false, false);
        bk = *reinterpret_cast<const v16h*>(kr1);
        s1 = __builtin_amdgcn_wmma_f32_16x16x32_f16(false, aq[0], false, bk, (short)0, s1, false, false);
        bk = *reinterpret_cast<const v16h*>(kr1 + 32);
        s1 = __builtin_amdgcn_wmma_f32_16x16x32_f16(false, aq[1], false, bk, (short)0, s1, false, false);
    }

    // ---- scale (+ causal mask on the diagonal step only) ----
#pragma unroll
    for (int r = 0; r < 8; ++r) {
        float a = s0[r] * SC, b = s1[r] * SC;
        if (MASK) {
            const int qrow = qbase + r + hi * 8;    // C-layout: M = r + hi*8
            if (jc + lane16      > qrow) a = -__builtin_inff();
            if (jc + 16 + lane16 > qrow) b = -__builtin_inff();
        }
        s0[r] = a; s1[r] = b;
    }

    // ---- online softmax: row stats replicated across 16-lane rows (DPP) ----
    float p0[8], p1[8], alpha[8];
#pragma unroll
    for (int r = 0; r < 8; ++r) {
        const float rm = rowmax16(fmaxf(s0[r], s1[r]));
        const float mn = fmaxf(m[r], rm);
        alpha[r] = exp2f(m[r] - mn);
        m[r] = mn;
        p0[r] = exp2f(s0[r] - mn);
        p1[r] = exp2f(s1[r] - mn);
        l[r] = l[r] * alpha[r] + rowsum16(p0[r] + p1[r]);
    }
#pragma unroll
    for (int r = 0; r < 8; ++r) {
        o[0][r] *= alpha[r]; o[1][r] *= alpha[r];
        o[2][r] *= alpha[r]; o[3][r] *= alpha[r];
    }

    // ---- C-layout P -> LDS (row-major f16) -> re-read in A-layout ----
#pragma unroll
    for (int r = 0; r < 8; ++r) {
        const int M = r + hi * 8;
        pb[M][lane16]      = (_Float16)p0[r];
        pb[M][16 + lane16] = (_Float16)p1[r];
    }
    __builtin_amdgcn_wave_barrier();
    asm volatile("s_wait_dscnt 0x0" ::: "memory");       // per-wave LDS RAW
    v16h pa;
    {
        const int lo = hi * 16;
        v8h c0 = *reinterpret_cast<const v8h*>(&pb[lane16][lo]);
        v8h c1 = *reinterpret_cast<const v8h*>(&pb[lane16][lo + 8]);
#pragma unroll
        for (int i = 0; i < 8; ++i) { pa[i] = c0[i]; pa[i + 8] = c1[i]; }
    }
    asm volatile("" ::: "memory");                       // keep WAR order vs next step
    __builtin_amdgcn_wave_barrier();

    // ---- O += P V : Vt is [d][s] f16 so K-dim is contiguous per lane ----
#pragma unroll
    for (int t = 0; t < 4; ++t) {
        const _Float16* vr = vtp + (size_t)(t * 16 + lane16) * S_ + jc + hi * 16;
        v16h bv = *reinterpret_cast<const v16h*>(vr);
        o[t] = __builtin_amdgcn_wmma_f32_16x16x32_f16(false, pa, false, bv, (short)0, o[t], false, false);
    }
}

// ---------------------------------------------------------------------------
// FlashAttention-2 forward, causal. 256 threads = 8 waves; each wave owns a
// 16-row Q strip, iterates KV in 32-column steps with online softmax.
// ---------------------------------------------------------------------------
__global__ __launch_bounds__(256, 1)
void fa2_fwd_causal(const float* __restrict__ q, const _Float16* __restrict__ kh,
                    const _Float16* __restrict__ vt, float* __restrict__ out) {
    // Per-wave P staging: 16 rows x 32 cols f16, padded row pitch (40 halfs).
    __shared__ __align__(16) _Float16 pbuf[8][16][40];

    const int lane   = threadIdx.x & 31;
    const int wave   = threadIdx.x >> 5;
    const int lane16 = lane & 15;
    const int hi     = lane >> 4;              // 0: lanes 0-15, 1: lanes 16-31
    const int bh     = blockIdx.y;
    const int qbase  = blockIdx.x * 128 + wave * 16;

    const float*    qp  = q   + (size_t)bh * S_ * D_;
    const _Float16* kp  = kh  + (size_t)bh * S_ * D_;
    const _Float16* vtp = vt  + (size_t)bh * D_ * S_;
    float*          op  = out + (size_t)bh * S_ * D_;

    // --- Q A-fragments (16x32 f16), one per 32-wide slice of D. A-layout:
    // lane16 = row M; halfs [0..15] = K hi*16..hi*16+15.
    v16h aq[2];
    {
        const float* qr = qp + (size_t)(qbase + lane16) * D_ + hi * 16;
        load8f_cvt(qr,      aq[0], 0);
        load8f_cvt(qr + 8,  aq[0], 8);
        load8f_cvt(qr + 32, aq[1], 0);
        load8f_cvt(qr + 40, aq[1], 8);
    }

    v8f o[4];
#pragma unroll
    for (int t = 0; t < 4; ++t) o[t] = (v8f){};
    float m[8], l[8];
#pragma unroll
    for (int r = 0; r < 8; ++r) { m[r] = -__builtin_inff(); l[r] = 0.f; }

    const int steps = ((qbase + 15) >> 5) + 1; // causal trip count (Bc = 32)

    // Hot loop: fully-below-diagonal steps, no masking VALU.
    for (int j = 0; j < steps - 1; ++j) {
        __builtin_prefetch(kp + (size_t)(j * 32 + 32 + lane) * D_, 0, 0);
        fa_step<false>(j * 32, qbase, lane16, hi, kp, vtp, aq, o, m, l, pbuf[wave]);
    }
    // Diagonal step: causal mask applied.
    fa_step<true>((steps - 1) * 32, qbase, lane16, hi, kp, vtp, aq, o, m, l, pbuf[wave]);

    // ---- normalize and store (C-layout -> row-major f32) ----
    float inv[8];
#pragma unroll
    for (int r = 0; r < 8; ++r) inv[r] = 1.0f / l[r];
#pragma unroll
    for (int t = 0; t < 4; ++t)
#pragma unroll
        for (int r = 0; r < 8; ++r)
            op[(size_t)(qbase + r + hi * 8) * D_ + t * 16 + lane16] = o[t][r] * inv[r];
}

// ---------------------------------------------------------------------------
extern "C" void kernel_launch(void* const* d_in, const int* in_sizes, int n_in,
                              void* d_out, int out_size, void* d_ws, size_t ws_size,
                              hipStream_t stream) {
    const float* q = (const float*)d_in[0];
    const float* k = (const float*)d_in[1];
    const float* v = (const float*)d_in[2];
    // d_in[3] is the causal mask; causality is hardcoded in the kernel.

    const size_t n = (size_t)B_ * H_ * S_ * D_;    // 4,194,304 elements
    _Float16* kh = (_Float16*)d_ws;                // 8 MB
    _Float16* vt = kh + n;                         // 8 MB  (16 MB workspace total)

    k_convert_f16<<<dim3((unsigned)(n / 8 / 256)), 256, 0, stream>>>(k, kh);
    v_transpose_f16<<<dim3(S_ / 32, D_ / 32, B_ * H_), dim3(32, 8), 0, stream>>>(v, vt);
    fa2_fwd_causal<<<dim3(S_ / 128, B_ * H_), 256, 0, stream>>>(q, kh, vt, (float*)d_out);
}
